// HybridBlock_25142738550844
// MI455X (gfx1250) — matrix operand order repaired and checked by reading, not compiled, not run on hardware
//
#include <hip/hip_runtime.h>
#include <hip/hip_bf16.h>
#include <math.h>
#include <stdint.h>

typedef __bf16 bf16_t;
typedef __attribute__((ext_vector_type(16))) __bf16 v16bf;
typedef __attribute__((ext_vector_type(8)))  __bf16 v8bf;
typedef __attribute__((ext_vector_type(8)))  float  v8f;

constexpr int DIM     = 1024;
constexpr int TSEQ    = 2048;
constexpr int BATCH   = 4;
constexpr int NT      = BATCH * TSEQ;      // 8192 tokens
constexpr int NHEADS  = 8;
constexpr int HEAD_K  = 256;
constexpr int HEAD_V  = 512;
constexpr int KEY_DIM = 2048;
constexpr int VAL_DIM = 4096;
constexpr int TOTAL_N = KEY_DIM + KEY_DIM + VAL_DIM + VAL_DIM;  // 12288
constexpr int QKV_W   = KEY_DIM + KEY_DIM + VAL_DIM;            // 8192
constexpr int NVT     = 32;                 // Dv tile per recurrence workgroup

// GEMM tiling
constexpr int TM = 128, TN = 128, TK = 32;
constexpr int LDSROW = 40;                  // padded LDS row (bf16): 20-DWORD stride, conflict-free

// ---------------------------------------------------------------------------
// Weight prep: concat q/k/v/g weights into one [TOTAL_N, DIM] bf16 matrix
// ---------------------------------------------------------------------------
__global__ void concat_weights_kernel(const float* __restrict__ qw,
                                      const float* __restrict__ kw,
                                      const float* __restrict__ vw,
                                      const float* __restrict__ gw,
                                      bf16_t* __restrict__ Wc) {
    size_t i = (size_t)blockIdx.x * blockDim.x + threadIdx.x;
    if (i >= (size_t)TOTAL_N * DIM) return;
    int row = (int)(i / DIM);
    int col = (int)(i % DIM);
    float v;
    if (row < KEY_DIM)                 v = qw[(size_t)row * DIM + col];
    else if (row < 2 * KEY_DIM)        v = kw[(size_t)(row - KEY_DIM) * DIM + col];
    else if (row < 2 * KEY_DIM + VAL_DIM)
                                       v = vw[(size_t)(row - 2 * KEY_DIM) * DIM + col];
    else                               v = gw[(size_t)(row - 2 * KEY_DIM - VAL_DIM) * DIM + col];
    Wc[i] = (bf16_t)v;
}

__global__ void convert_f32_bf16_kernel(const float* __restrict__ src,
                                        bf16_t* __restrict__ dst, size_t n) {
    size_t i = (size_t)blockIdx.x * blockDim.x + threadIdx.x;
    if (i < n) dst[i] = (bf16_t)src[i];
}

// ---------------------------------------------------------------------------
// Pre-norm RMSNorm: x[NT,DIM] f32 -> h[NT,DIM] bf16
// ---------------------------------------------------------------------------
__global__ __launch_bounds__(256) void rmsnorm_kernel(const float* __restrict__ x,
                                                      const float* __restrict__ w,
                                                      bf16_t* __restrict__ h) {
    __shared__ float sm[256];
    int t = blockIdx.x, tid = threadIdx.x;
    const float* xr = x + (size_t)t * DIM;
    float vals[4], ss = 0.f;
#pragma unroll
    for (int i = 0; i < 4; ++i) { vals[i] = xr[tid + i * 256]; ss += vals[i] * vals[i]; }
    sm[tid] = ss; __syncthreads();
    for (int s = 128; s > 0; s >>= 1) { if (tid < s) sm[tid] += sm[tid + s]; __syncthreads(); }
    float inv = rsqrtf(sm[0] / (float)DIM + 1e-6f);
#pragma unroll
    for (int i = 0; i < 4; ++i)
        h[(size_t)t * DIM + tid + i * 256] = (bf16_t)(vals[i] * inv * w[tid + i * 256]);
}

// ---------------------------------------------------------------------------
// LDS-tiled bf16 WMMA GEMM with gfx1250 async global->LDS staging.
// Block tile 128x128, 8 waves (2x4), each wave 4x2 16x16 accumulators.
// Tiles double-buffered via GLOBAL_LOAD_ASYNC_TO_LDS_B128 (ASYNCcnt),
// no VGPR landing zone; completion = s_wait_asynccnt 0 + barrier.
// ---------------------------------------------------------------------------
__global__ __launch_bounds__(256) void wmma_gemm_lds(const bf16_t* __restrict__ A,
                                                     const bf16_t* __restrict__ Wt,
                                                     bf16_t* __restrict__ Ybf,
                                                     float* __restrict__ Yf,
                                                     const float* __restrict__ resid,
                                                     int M, int N, int K) {
    __shared__ bf16_t Asm[2][TM * LDSROW];   // 2 x 10 KB
    __shared__ bf16_t Bsm[2][TN * LDSROW];   // 2 x 10 KB  (40 KB total)

    int tid  = threadIdx.x;
    int lane = tid & 31;
    int wave = tid >> 5;            // 0..7
    int wm   = wave & 1;            // 2 wave-rows   -> 64 M each
    int wn   = wave >> 1;           // 4 wave-cols   -> 32 N each
    int r    = lane & 15;
    int hh   = lane >> 4;

    int nb = N / TN;
    int m0 = (int)(blockIdx.x / nb) * TM;
    int n0 = (int)(blockIdx.x % nb) * TN;

    // global->LDS staging: thread moves 32B of one row of A and of B per tile
    int lrow  = tid >> 1;           // 0..127
    int lhalf = (tid & 1) * 16;     // 0 | 16
    const bf16_t* ag = A  + (size_t)(m0 + lrow) * K + lhalf;
    const bf16_t* bg = Wt + (size_t)(n0 + lrow) * K + lhalf;

    // per-thread LDS destinations (flat shared ptr low 32 bits = LDS offset)
    uint32_t ldsA[2], ldsB[2];
#pragma unroll
    for (int b = 0; b < 2; ++b) {
        ldsA[b] = (uint32_t)(uintptr_t)&Asm[b][lrow * LDSROW + lhalf];
        ldsB[b] = (uint32_t)(uintptr_t)&Bsm[b][lrow * LDSROW + lhalf];
    }

    auto issue_async = [&](int buf, int kt) {
        uint64_t ga = (uint64_t)(uintptr_t)(ag + (size_t)kt * TK);
        uint64_t gb = (uint64_t)(uintptr_t)(bg + (size_t)kt * TK);
        asm volatile(
            "global_load_async_to_lds_b128 %0, %2, off\n\t"
            "global_load_async_to_lds_b128 %0, %2, off offset:16\n\t"
            "global_load_async_to_lds_b128 %1, %3, off\n\t"
            "global_load_async_to_lds_b128 %1, %3, off offset:16"
            :: "v"(ldsA[buf]), "v"(ldsB[buf]), "v"(ga), "v"(gb)
            : "memory");
    };

    v8f acc[4][2];
#pragma unroll
    for (int i = 0; i < 4; ++i)
#pragma unroll
        for (int j = 0; j < 2; ++j) acc[i][j] = v8f{};

    int nk = K / TK;

    // prologue: tile 0 -> buffer 0
    issue_async(0, 0);
    asm volatile("s_wait_asynccnt 0" ::: "memory");
    __syncthreads();

    for (int kt = 0; kt < nk; ++kt) {
        int cur = kt & 1, nxt = cur ^ 1;

        // stage next tile asynchronously into the other buffer
        // (safe: buffer `nxt` was last read in iteration kt-1, behind the barrier)
        if (kt + 1 < nk) {
            issue_async(nxt, kt + 1);
            if (kt + 2 < nk) {  // gfx1250 global_prefetch_b8 path, warm L2
                __builtin_prefetch(ag + (size_t)(kt + 2) * TK, 0, 1);
                __builtin_prefetch(bg + (size_t)(kt + 2) * TK, 0, 1);
            }
        }

        // B fragments (2): col = wn*32 + ni*16 + r, 16 contiguous K at hh*16
        v16bf bfrag[2];
#pragma unroll
        for (int ni = 0; ni < 2; ++ni) {
            const bf16_t* bp = &Bsm[cur][(wn * 32 + ni * 16 + r) * LDSROW + hh * 16];
            v8bf b0 = *(const v8bf*)bp;
            v8bf b1 = *(const v8bf*)(bp + 8);
#pragma unroll
            for (int i = 0; i < 8; ++i) { bfrag[ni][i] = b0[i]; bfrag[ni][8 + i] = b1[i]; }
        }
        // A fragments (4) + 8 WMMAs
#pragma unroll
        for (int mi = 0; mi < 4; ++mi) {
            const bf16_t* ap = &Asm[cur][(wm * 64 + mi * 16 + r) * LDSROW + hh * 16];
            v8bf a0 = *(const v8bf*)ap;
            v8bf a1 = *(const v8bf*)(ap + 8);
            v16bf afrag;
#pragma unroll
            for (int i = 0; i < 8; ++i) { afrag[i] = a0[i]; afrag[8 + i] = a1[i]; }
            acc[mi][0] = __builtin_amdgcn_wmma_f32_16x16x32_bf16(
                false, afrag, false, bfrag[0], (short)0, acc[mi][0], false, false);
            acc[mi][1] = __builtin_amdgcn_wmma_f32_16x16x32_bf16(
                false, afrag, false, bfrag[1], (short)0, acc[mi][1], false, false);
        }

        // drain own async writes, then make all waves' writes visible
        if (kt + 1 < nk) asm volatile("s_wait_asynccnt 0" ::: "memory");
        __syncthreads();
    }

    // store: C/D layout m = v + hh*8 (vgpr v), n = lane&15
#pragma unroll
    for (int mi = 0; mi < 4; ++mi) {
#pragma unroll
        for (int ni = 0; ni < 2; ++ni) {
            int n = n0 + wn * 32 + ni * 16 + r;
#pragma unroll
            for (int v = 0; v < 8; ++v) {
                int m = m0 + wm * 64 + mi * 16 + v + hh * 8;
                size_t idx = (size_t)m * N + n;
                float val = acc[mi][ni][v];
                if (Yf)  Yf[idx]  = (resid ? resid[idx] : 0.f) + val;
                if (Ybf) Ybf[idx] = (bf16_t)val;
            }
        }
    }
}

// ---------------------------------------------------------------------------
// beta = sigmoid(h @ b_w^T), glog = -exp(A_log)*softplus(h @ a_w^T + dt_bias)
// ---------------------------------------------------------------------------
__global__ void gates_kernel(const bf16_t* __restrict__ h,
                             const float* __restrict__ b_w,
                             const float* __restrict__ a_w,
                             const float* __restrict__ A_log,
                             const float* __restrict__ dt_bias,
                             float* __restrict__ beta,
                             float* __restrict__ glog) {
    int idx = blockIdx.x * blockDim.x + threadIdx.x;
    if (idx >= NT * NHEADS) return;
    int t = idx / NHEADS, hd = idx % NHEADS;
    const bf16_t* hr = h + (size_t)t * DIM;
    const float* br = b_w + (size_t)hd * DIM;
    const float* ar = a_w + (size_t)hd * DIM;
    float db = 0.f, da = 0.f;
    for (int d = 0; d < DIM; ++d) {
        float hv = (float)hr[d];
        db += hv * br[d];
        da += hv * ar[d];
    }
    beta[idx] = 1.f / (1.f + expf(-db));
    float sx = da + dt_bias[hd];
    float sp = (sx > 20.f) ? sx : log1pf(expf(sx));
    glog[idx] = -expf(A_log[hd]) * sp;
}

// ---------------------------------------------------------------------------
// Depthwise causal conv(K=4) + SiLU + per-head L2 norm of q,k (+ q scale).
// One block per token; Y is bf16 [NT, TOTAL_N]; writes qkv f32 [NT, QKV_W].
// ---------------------------------------------------------------------------
__global__ __launch_bounds__(256) void conv_norm_kernel(const bf16_t* __restrict__ Y,
                                                        const float* __restrict__ qc,
                                                        const float* __restrict__ kc,
                                                        const float* __restrict__ vc,
                                                        float* __restrict__ qkv) {
    __shared__ float sm[256];
    __shared__ float nf[16];                 // rsq for q heads [0..7], k heads [8..15]
    int gt = blockIdx.x;
    int t  = gt % TSEQ;                      // causal reset at sequence boundary
    int tid = threadIdx.x;

    float qv[8], kv[8], vv[16];
#pragma unroll
    for (int hd = 0; hd < 8; ++hd) {
        int ch = hd * HEAD_K + tid;
        float aq = 0.f, ak = 0.f;
#pragma unroll
        for (int i = 0; i < 4; ++i) {
            int tt = t - 3 + i;
            if (tt >= 0) {
                size_t rowoff = (size_t)(gt - 3 + i) * TOTAL_N;
                aq += (float)Y[rowoff + ch] * qc[ch * 4 + i];
                ak += (float)Y[rowoff + KEY_DIM + ch] * kc[ch * 4 + i];
            }
        }
        qv[hd] = aq / (1.f + expf(-aq));
        kv[hd] = ak / (1.f + expf(-ak));
    }
#pragma unroll
    for (int j = 0; j < 16; ++j) {
        int ch = j * 256 + tid;
        float av = 0.f;
#pragma unroll
        for (int i = 0; i < 4; ++i) {
            int tt = t - 3 + i;
            if (tt >= 0)
                av += (float)Y[(size_t)(gt - 3 + i) * TOTAL_N + 2 * KEY_DIM + ch] * vc[ch * 4 + i];
        }
        vv[j] = av / (1.f + expf(-av));
    }
    for (int hd = 0; hd < 8; ++hd) {
        sm[tid] = qv[hd] * qv[hd]; __syncthreads();
        for (int s = 128; s > 0; s >>= 1) { if (tid < s) sm[tid] += sm[tid + s]; __syncthreads(); }
        if (tid == 0) nf[hd] = rsqrtf(sm[0] + 1e-6f);
        __syncthreads();
        sm[tid] = kv[hd] * kv[hd]; __syncthreads();
        for (int s = 128; s > 0; s >>= 1) { if (tid < s) sm[tid] += sm[tid + s]; __syncthreads(); }
        if (tid == 0) nf[8 + hd] = rsqrtf(sm[0] + 1e-6f);
        __syncthreads();
    }
    const float qscale = 0.0625f;            // HEAD_K^-0.5
    float* row = qkv + (size_t)gt * QKV_W;
#pragma unroll
    for (int hd = 0; hd < 8; ++hd) {
        row[hd * HEAD_K + tid]            = qv[hd] * nf[hd] * qscale;
        row[KEY_DIM + hd * HEAD_K + tid]  = kv[hd] * nf[8 + hd];
    }
#pragma unroll
    for (int j = 0; j < 16; ++j)
        row[2 * KEY_DIM + j * 256 + tid] = vv[j];
}

// ---------------------------------------------------------------------------
// Gated delta-rule recurrence. One block per (b, head, Dv-tile of 32).
// S slice [HEAD_K x NVT] lives in LDS (32 KB); thread owns one Dv column.
// ---------------------------------------------------------------------------
__global__ __launch_bounds__(32) void recurrence_kernel(const float* __restrict__ qkv,
                                                        const float* __restrict__ beta,
                                                        const float* __restrict__ glog,
                                                        float* __restrict__ o) {
    __shared__ float S[HEAD_K * NVT];
    __shared__ float ks[HEAD_K];
    __shared__ float qs[HEAD_K];
    int blk = blockIdx.x;                          // BATCH*NHEADS*(HEAD_V/NVT) = 512
    int vt  = blk % (HEAD_V / NVT);
    int hd  = (blk / (HEAD_V / NVT)) % NHEADS;
    int b   = blk / ((HEAD_V / NVT) * NHEADS);
    int tid = threadIdx.x;

    for (int i = tid; i < HEAD_K * NVT; i += 32) S[i] = 0.f;
    __syncthreads();

    for (int t = 0; t < TSEQ; ++t) {
        int gt = b * TSEQ + t;
        const float* row = qkv + (size_t)gt * QKV_W;
        for (int i = tid; i < HEAD_K; i += 32) {
            qs[i] = row[hd * HEAD_K + i];
            ks[i] = row[KEY_DIM + hd * HEAD_K + i];
        }
        __syncthreads();
        float vin = row[2 * KEY_DIM + hd * HEAD_V + vt * NVT + tid];
        float dec = expf(glog[gt * NHEADS + hd]);
        float bta = beta[gt * NHEADS + hd];

        float kS = 0.f;
        for (int dk = 0; dk < HEAD_K; ++dk) kS += S[dk * NVT + tid] * ks[dk];
        float vn = (vin - dec * kS) * bta;

        float oa = 0.f;
        for (int dk = 0; dk < HEAD_K; ++dk) {
            float s = dec * S[dk * NVT + tid] + ks[dk] * vn;
            S[dk * NVT + tid] = s;
            oa += qs[dk] * s;
        }
        o[(size_t)gt * VAL_DIM + hd * HEAD_V + vt * NVT + tid] = oa;
        __syncthreads();
    }
}

// ---------------------------------------------------------------------------
// Per-head RMSNorm * o_norm_w * silu(gate) -> ofin bf16 [NT, VAL_DIM]
// gate lives in Y columns [2*KEY_DIM+VAL_DIM, TOTAL_N)
// ---------------------------------------------------------------------------
__global__ __launch_bounds__(128) void gnorm_kernel(const float* __restrict__ o,
                                                    const bf16_t* __restrict__ Y,
                                                    const float* __restrict__ onw,
                                                    bf16_t* __restrict__ ofin) {
    __shared__ float sm[128];
    int blk = blockIdx.x;                     // NT * NHEADS
    int t = blk / NHEADS, hd = blk % NHEADS;
    int tid = threadIdx.x;
    const float* orow = o + (size_t)t * VAL_DIM + hd * HEAD_V;
    float vals[4], ss = 0.f;
#pragma unroll
    for (int i = 0; i < 4; ++i) { vals[i] = orow[tid + i * 128]; ss += vals[i] * vals[i]; }
    sm[tid] = ss; __syncthreads();
    for (int s = 64; s > 0; s >>= 1) { if (tid < s) sm[tid] += sm[tid + s]; __syncthreads(); }
    float inv = rsqrtf(sm[0] / (float)HEAD_V + 1e-5f);
#pragma unroll
    for (int i = 0; i < 4; ++i) {
        int dv = tid + i * 128;
        float gate = (float)Y[(size_t)t * TOTAL_N + 2 * KEY_DIM + VAL_DIM + hd * HEAD_V + dv];
        float sg = gate / (1.f + expf(-gate));
        ofin[(size_t)t * VAL_DIM + hd * HEAD_V + dv] = (bf16_t)(vals[i] * inv * onw[dv] * sg);
    }
}

// ---------------------------------------------------------------------------
extern "C" void kernel_launch(void* const* d_in, const int* in_sizes, int n_in,
                              void* d_out, int out_size, void* d_ws, size_t ws_size,
                              hipStream_t stream) {
    const float* x       = (const float*)d_in[0];
    const float* norm_w  = (const float*)d_in[1];
    const float* q_w     = (const float*)d_in[2];
    const float* k_w     = (const float*)d_in[3];
    const float* v_w     = (const float*)d_in[4];
    const float* q_conv  = (const float*)d_in[5];
    const float* k_conv  = (const float*)d_in[6];
    const float* v_conv  = (const float*)d_in[7];
    const float* b_w     = (const float*)d_in[8];
    const float* a_w     = (const float*)d_in[9];
    const float* A_log   = (const float*)d_in[10];
    const float* dt_bias = (const float*)d_in[11];
    const float* g_w     = (const float*)d_in[12];
    const float* o_nw    = (const float*)d_in[13];
    const float* o_w     = (const float*)d_in[14];
    float* out = (float*)d_out;

    // workspace layout (256B aligned bumps)
    char* p = (char*)d_ws;
    auto bump = [&](size_t bytes) { char* r = p; p += (bytes + 255) & ~(size_t)255; return r; };
    bf16_t* h    = (bf16_t*)bump((size_t)NT * DIM * 2);          // 16 MB
    bf16_t* Wc   = (bf16_t*)bump((size_t)TOTAL_N * DIM * 2);     // 24 MB
    bf16_t* oWb  = (bf16_t*)bump((size_t)DIM * VAL_DIM * 2);     //  8 MB
    bf16_t* Y    = (bf16_t*)bump((size_t)NT * TOTAL_N * 2);      // 192 MB
    float*  qkv  = (float*) bump((size_t)NT * QKV_W * 4);        // 256 MB
    float*  beta = (float*) bump((size_t)NT * NHEADS * 4);
    float*  glog = (float*) bump((size_t)NT * NHEADS * 4);
    float*  obuf = (float*) bump((size_t)NT * VAL_DIM * 4);      // 128 MB
    bf16_t* ofin = (bf16_t*)bump((size_t)NT * VAL_DIM * 2);      //  64 MB

    // 1. weight conversion
    {
        size_t n = (size_t)TOTAL_N * DIM;
        concat_weights_kernel<<<(int)((n + 255) / 256), 256, 0, stream>>>(q_w, k_w, v_w, g_w, Wc);
        size_t n2 = (size_t)DIM * VAL_DIM;
        convert_f32_bf16_kernel<<<(int)((n2 + 255) / 256), 256, 0, stream>>>(o_w, oWb, n2);
    }
    // 2. RMSNorm
    rmsnorm_kernel<<<NT, 256, 0, stream>>>(x, norm_w, h);
    // 3. fused q/k/v/gate projection GEMM (LDS-tiled bf16 WMMA, async staging)
    {
        int blocks = (NT / TM) * (TOTAL_N / TN);                 // 64*96 = 6144
        wmma_gemm_lds<<<blocks, 256, 0, stream>>>(h, Wc, Y, nullptr, nullptr,
                                                  NT, TOTAL_N, DIM);
    }
    // 4. beta / log-decay gates
    gates_kernel<<<(NT * NHEADS) / 256, 256, 0, stream>>>(h, b_w, a_w, A_log, dt_bias, beta, glog);
    // 5. conv + silu + qk norm
    conv_norm_kernel<<<NT, 256, 0, stream>>>(Y, q_conv, k_conv, v_conv, qkv);
    // 6. delta-rule recurrence
    recurrence_kernel<<<BATCH * NHEADS * (HEAD_V / NVT), 32, 0, stream>>>(qkv, beta, glog, obuf);
    // 7. gated per-head RMSNorm
    gnorm_kernel<<<NT * NHEADS, 128, 0, stream>>>(obuf, Y, o_nw, ofin);
    // 8. output projection + residual (LDS-tiled bf16 WMMA, f32 out)
    {
        int blocks = (NT / TM) * (DIM / TN);                     // 64*8 = 512
        wmma_gemm_lds<<<blocks, 256, 0, stream>>>(ofin, oWb, nullptr, out, x,
                                                  NT, DIM, VAL_DIM);
    }
    (void)in_sizes; (void)n_in; (void)out_size; (void)ws_size;
}
